// MultiSelfAttention_38079180047148
// MI455X (gfx1250) — matrix operand order repaired and checked by reading, not compiled, not run on hardware
//
#include <hip/hip_runtime.h>

// ---------------------------------------------------------------------------
// Problem constants: B=2, S=2048, D=512, H=8, HID=512, DH=64
// ---------------------------------------------------------------------------
constexpr int kB   = 2;
constexpr int kS   = 2048;
constexpr int kD   = 512;
constexpr int kH   = 8;
constexpr int kHID = 512;
constexpr int kDH  = 64;
constexpr int kM   = kB * kS;        // 4096 rows for all GEMMs

typedef __attribute__((ext_vector_type(16))) __bf16 v16bf;
typedef __attribute__((ext_vector_type(8)))  float  v8f;

// f32 -> bf16, round-to-nearest-even
static __device__ __forceinline__ unsigned short f32_to_bf16(float f) {
  unsigned int u = __float_as_uint(f);
  u += 0x7FFFu + ((u >> 16) & 1u);
  return (unsigned short)(u >> 16);
}

struct U32x8 { uint4 lo, hi; };

static __device__ __forceinline__ v16bf make_frag(uint4 lo, uint4 hi) {
  U32x8 p; p.lo = lo; p.hi = hi;
  return __builtin_bit_cast(v16bf, p);
}
// Fragment = two 16-byte chunks (8 bf16 each)
static __device__ __forceinline__ v16bf load_frag(const unsigned short* p0,
                                                  const unsigned short* p1) {
  return make_frag(*(const uint4*)p0, *(const uint4*)p1);
}
static __device__ __forceinline__ v8f zero8() {
  v8f z = {0.f,0.f,0.f,0.f,0.f,0.f,0.f,0.f};
  return z;
}
static __device__ __forceinline__ v8f wmma_bf16(v16bf a, v16bf b, v8f c) {
  return __builtin_amdgcn_wmma_f32_16x16x32_bf16(false, a, false, b,
                                                 (short)0, c, false, false);
}

// LDS byte offset = low 32 bits of the generic pointer (amdgcn flat pointer to
// LDS is {shared_aperture_hi, as3_offset}).
static __device__ __forceinline__ unsigned int lds_addr32(const void* p) {
  return (unsigned int)(unsigned long long)(uintptr_t)p;
}
// Async DMA: 16 bytes global -> LDS per lane; tracked by ASYNCcnt.
static __device__ __forceinline__ void async_ld16(unsigned int lds_byte,
                                                  const void* gaddr) {
  asm volatile("global_load_async_to_lds_b128 %0, %1, off"
               :: "v"(lds_byte), "v"(gaddr) : "memory");
}
static __device__ __forceinline__ void wait_async0() {
  asm volatile("s_wait_asynccnt 0x0" ::: "memory");
}

// ---------------------------------------------------------------------------
// Kernel 0a: f32 -> bf16 elementwise convert
// ---------------------------------------------------------------------------
__global__ void __launch_bounds__(256)
cvt_bf16_kernel(const float* __restrict__ src, unsigned short* __restrict__ dst, int n) {
  int i = blockIdx.x * blockDim.x + threadIdx.x;
  if (i < n) dst[i] = f32_to_bf16(src[i]);
}

// ---------------------------------------------------------------------------
// Kernel 0b: transpose + convert weight [K,N] f32 -> [N,K] bf16
// ---------------------------------------------------------------------------
__global__ void __launch_bounds__(256)
transpose_cvt_kernel(const float* __restrict__ src, unsigned short* __restrict__ dst,
                     int rows, int cols) {
  int i = blockIdx.x * blockDim.x + threadIdx.x;
  if (i < rows * cols) {
    int k = i / cols;
    int n = i - k * cols;
    dst[(size_t)n * rows + k] = f32_to_bf16(src[i]);
  }
}

// ---------------------------------------------------------------------------
// WMMA GEMM: out = X(bf16,[M,K]) @ WT(bf16,[N,K] row-major == W^T) + bias
// B-tile (64x32) staged in LDS via async DMA, double buffered. Each wave
// computes a 32x64 output tile (2 A-frags share 4 B-frags -> 8 WMMA/k-step).
//   mode 0: bf16 row-major | mode 1: bf16 scatter to V^T head layout |
//   mode 2: f32 row-major
// Block = 256 threads (8 waves) covering 256x64 of the output.
// ---------------------------------------------------------------------------
__global__ void __launch_bounds__(256)
gemm_bf16_wmma(const unsigned short* __restrict__ X,
               const unsigned short* __restrict__ WT,
               const float* __restrict__ bias,
               void* __restrict__ out,
               int M, int N, int K, int mode) {
  __shared__ unsigned short ldsB[2][64 * 32];    // 2 x 4KB weight tiles

  const int tid  = threadIdx.x;
  const int w    = tid >> 5;
  const int lane = tid & 31;
  const int hf   = lane >> 4;
  const int ln   = lane & 15;
  const int mbase = blockIdx.x * 256 + w * 32;
  const int nbase = blockIdx.y * 64;

  // DMA mapping: 256 lanes x 16B = one 64x32 bf16 tile
  const int crow = tid >> 2;                     // 0..63 (tile row = local n)
  const int cchk = tid & 3;                      // 0..3  (16B chunk in row)
  const unsigned short* wsrc = WT + (size_t)(nbase + crow) * K + cchk * 8;
  const unsigned int ldsb0 = lds_addr32(&ldsB[0][0]) + crow * 64 + cchk * 16;
  const unsigned int ldsb1 = lds_addr32(&ldsB[1][0]) + crow * 64 + cchk * 16;

  const unsigned short* xr0 = X + (size_t)(mbase + ln) * K;
  const unsigned short* xr1 = X + (size_t)(mbase + 16 + ln) * K;
  v8f acc[2][4] = {{zero8(), zero8(), zero8(), zero8()},
                   {zero8(), zero8(), zero8(), zero8()}};

  // prologue: stage k-tile 0, preload first A fragments
  async_ld16(ldsb0, wsrc);
  v16bf a0 = load_frag(xr0 + hf * 8, xr0 + 16 + hf * 8);
  v16bf a1 = load_frag(xr1 + hf * 8, xr1 + 16 + hf * 8);
  wait_async0();
  __syncthreads();

  for (int kb = 0; kb < K; kb += 32) {
    const int  cur  = (kb >> 5) & 1;
    const bool more = (kb + 32) < K;
    // issue DMA for next tile; it overlaps this iteration's WMMAs
    if (more) async_ld16(cur ? ldsb0 : ldsb1, wsrc + kb + 32);
    // register double-buffer the A fragments
    v16bf a0n = a0, a1n = a1;
    if (more) {
      a0n = load_frag(xr0 + kb + 32 + hf * 8, xr0 + kb + 48 + hf * 8);
      a1n = load_frag(xr1 + kb + 32 + hf * 8, xr1 + kb + 48 + hf * 8);
    }

    // batch all 4 B-fragment LDS loads, then issue the 8 WMMAs
    const unsigned short* bt = &ldsB[cur][0];
    v16bf bfr[4];
#pragma unroll
    for (int nt = 0; nt < 4; ++nt) {
      const unsigned short* br = bt + (nt * 16 + ln) * 32 + hf * 8;
      bfr[nt] = load_frag(br, br + 16);
    }
#pragma unroll
    for (int nt = 0; nt < 4; ++nt) {
      acc[0][nt] = wmma_bf16(a0, bfr[nt], acc[0][nt]);
      acc[1][nt] = wmma_bf16(a1, bfr[nt], acc[1][nt]);
    }
    a0 = a0n; a1 = a1n;
    if (more) wait_async0();
    __syncthreads();
  }

#pragma unroll
  for (int mt = 0; mt < 2; ++mt) {
#pragma unroll
    for (int nt = 0; nt < 4; ++nt) {
      const int n  = nbase + nt * 16 + ln;
      const float bv = bias[n];
#pragma unroll
      for (int r = 0; r < 8; ++r) {
        const int m = mbase + mt * 16 + r + 8 * hf;  // C-layout row
        const float val = acc[mt][nt][r] + bv;
        if (mode == 2) {
          ((float*)out)[(size_t)m * N + n] = val;
        } else if (mode == 0) {
          ((unsigned short*)out)[(size_t)m * N + n] = f32_to_bf16(val);
        } else {
          // V^T head layout: vpT[((b*8+h)*64 + d) * 2048 + t]
          const int b    = m >> 11;
          const int s    = m & (kS - 1);
          const int h    = s >> 8;
          const int trow = ((s & 255) << 3) + (n >> 6);
          const int d    = n & 63;
          ((unsigned short*)out)[(((size_t)(b * kH + h) * kDH + d) << 11) + trow] =
              f32_to_bf16(val);
        }
      }
    }
  }
}

// ---------------------------------------------------------------------------
// Flash attention per head-block. qp/kp flat bf16 (head = contiguous 2048x64),
// vpT = per-head transposed V [64 x 2048]. Output: concat[b, t, h*64+d] bf16.
// Shared K-tile (32x64) and V^T-tile (64x32) staged in LDS via async DMA,
// double buffered; 8 waves, each owns 16 query rows (flash online softmax).
// ---------------------------------------------------------------------------
__global__ void __launch_bounds__(256)
attn_wmma_kernel(const unsigned short* __restrict__ qp,
                 const unsigned short* __restrict__ kp,
                 const unsigned short* __restrict__ vpT,
                 unsigned short* __restrict__ concat) {
  __shared__ unsigned short ldsK[2][32 * 64];    // 2 x 4KB
  __shared__ unsigned short ldsV[2][64 * 32];    // 2 x 4KB
  __shared__ unsigned short ldsP[8][16 * 32];    // per-wave P tiles

  const int tid  = threadIdx.x;
  const int w    = tid >> 5;
  const int lane = tid & 31;
  const int hf   = lane >> 4;
  const int ln   = lane & 15;
  const int head = blockIdx.y;                   // 0..15 == b*8+h
  const int b    = head >> 3;
  const int h    = head & 7;
  const size_t hb = (size_t)head * kS * kDH;
  const int q0 = blockIdx.x * 128 + w * 16;

  // DMA mappings (256 lanes x 16B per tile)
  const int krow = tid >> 3, kchk = tid & 7;     // K tile: 32 rows x 128B
  const int vrow = tid >> 2, vchk = tid & 3;     // V tile: 64 rows x 64B
  const unsigned short* ksrc = kp  + hb + (size_t)krow * kDH + kchk * 8;
  const unsigned short* vsrc = vpT + hb + ((size_t)vrow << 11) + vchk * 8;
  const unsigned int ldsk0 = lds_addr32(&ldsK[0][0]) + krow * 128 + kchk * 16;
  const unsigned int ldsk1 = lds_addr32(&ldsK[1][0]) + krow * 128 + kchk * 16;
  const unsigned int ldsv0 = lds_addr32(&ldsV[0][0]) + vrow * 64 + vchk * 16;
  const unsigned int ldsv1 = lds_addr32(&ldsV[1][0]) + vrow * 64 + vchk * 16;

  // Q A-fragments for DH=64 (two K=32 fragments), resident for the whole loop
  const unsigned short* qr = qp + hb + (size_t)(q0 + ln) * kDH;
  const v16bf qa0 = load_frag(qr + hf * 8,      qr + 16 + hf * 8);
  const v16bf qa1 = load_frag(qr + 32 + hf * 8, qr + 48 + hf * 8);

  v8f acc[4] = {zero8(), zero8(), zero8(), zero8()};
  float mrow[8], lrow[8];
#pragma unroll
  for (int r = 0; r < 8; ++r) { mrow[r] = -1e30f; lrow[r] = 0.f; }

  unsigned short* pbuf = &ldsP[w][0];

  // prologue: stage kv tile 0
  async_ld16(ldsk0, ksrc);
  async_ld16(ldsv0, vsrc);
  wait_async0();
  __syncthreads();

  for (int kv = 0; kv < kS; kv += 32) {
    const int  cur  = (kv >> 5) & 1;
    const bool more = (kv + 32) < kS;
    if (more) {
      async_ld16(cur ? ldsk0 : ldsk1, ksrc + (size_t)(kv + 32) * kDH);
      async_ld16(cur ? ldsv0 : ldsv1, vsrc + kv + 32);
      if (kv + 64 < kS) __builtin_prefetch(ksrc + (size_t)(kv + 64) * kDH, 0, 0);
    }

    const unsigned short* kt = &ldsK[cur][0];
    const unsigned short* vt = &ldsV[cur][0];

    // ---- scores: batch all 4 K-fragment loads, then 4 WMMAs ----
    v8f c0 = zero8(), c1 = zero8();
    {
      const unsigned short* k0 = kt + ln * 64 + hf * 8;
      const unsigned short* k1 = kt + (16 + ln) * 64 + hf * 8;
      v16bf kb00 = load_frag(k0,      k0 + 16);
      v16bf kb01 = load_frag(k0 + 32, k0 + 48);
      v16bf kb10 = load_frag(k1,      k1 + 16);
      v16bf kb11 = load_frag(k1 + 32, k1 + 48);
      c0 = wmma_bf16(qa0, kb00, c0);
      c0 = wmma_bf16(qa1, kb01, c0);
      c1 = wmma_bf16(qa0, kb10, c1);
      c1 = wmma_bf16(qa1, kb11, c1);
    }

    // ---- online softmax: row stats replicated across each 16-lane half ----
#pragma unroll
    for (int r = 0; r < 8; ++r) {
      float v0 = c0[r], v1 = c1[r];
      float mx = fmaxf(v0, v1);
      mx = fmaxf(mx, __shfl_xor(mx, 1, 16));
      mx = fmaxf(mx, __shfl_xor(mx, 2, 16));
      mx = fmaxf(mx, __shfl_xor(mx, 4, 16));
      mx = fmaxf(mx, __shfl_xor(mx, 8, 16));
      float mnew = fmaxf(mrow[r], mx);
      float sc   = __expf(mrow[r] - mnew);
      mrow[r] = mnew;
      float p0 = __expf(v0 - mnew);
      float p1 = __expf(v1 - mnew);
      float ps = p0 + p1;
      ps += __shfl_xor(ps, 1, 16);
      ps += __shfl_xor(ps, 2, 16);
      ps += __shfl_xor(ps, 4, 16);
      ps += __shfl_xor(ps, 8, 16);
      lrow[r] = lrow[r] * sc + ps;
#pragma unroll
      for (int nt = 0; nt < 4; ++nt) acc[nt][r] *= sc;
      // C-layout (row r+8*hf, col ln / 16+ln) -> per-wave LDS P-tile 16x32
      pbuf[(r + 8 * hf) * 32 + ln]      = f32_to_bf16(p0);
      pbuf[(r + 8 * hf) * 32 + 16 + ln] = f32_to_bf16(p1);
    }

    // ---- re-read P as A-fragment (same wave: DScnt ordering suffices) ----
    const unsigned short* pr = pbuf + ln * 32;
    v16bf pa = load_frag(pr + hf * 8, pr + 16 + hf * 8);

    // ---- ctx += P(16x32) @ V(32x64): batch V-fragment loads, then WMMAs ----
    v16bf vb[4];
#pragma unroll
    for (int nt = 0; nt < 4; ++nt) {
      const unsigned short* vr = vt + (nt * 16 + ln) * 32 + hf * 8;
      vb[nt] = load_frag(vr, vr + 16);
    }
#pragma unroll
    for (int nt = 0; nt < 4; ++nt) {
      acc[nt] = wmma_bf16(pa, vb[nt], acc[nt]);
    }

    if (more) wait_async0();
    __syncthreads();
  }

  // ---- epilogue: divide by rowsum and sqrt(DH)=8 (applied AFTER softmax) ----
#pragma unroll
  for (int nt = 0; nt < 4; ++nt) {
    const int d = nt * 16 + ln;
#pragma unroll
    for (int r = 0; r < 8; ++r) {
      const int row = q0 + r + 8 * hf;
      const float val = acc[nt][r] / (lrow[r] * 8.0f);
      concat[((size_t)b * kS + row) * kHID + h * kDH + d] = f32_to_bf16(val);
    }
  }
}

// ---------------------------------------------------------------------------
// Host-side launch
// ---------------------------------------------------------------------------
extern "C" void kernel_launch(void* const* d_in, const int* in_sizes, int n_in,
                              void* d_out, int out_size, void* d_ws, size_t ws_size,
                              hipStream_t stream) {
  const float* q  = (const float*)d_in[0];
  const float* k  = (const float*)d_in[1];
  const float* v  = (const float*)d_in[2];
  const float* Wq = (const float*)d_in[3];
  const float* bq = (const float*)d_in[4];
  const float* Wk = (const float*)d_in[5];
  const float* bk = (const float*)d_in[6];
  const float* Wv = (const float*)d_in[7];
  const float* bv = (const float*)d_in[8];
  const float* Wo = (const float*)d_in[9];
  const float* bo = (const float*)d_in[10];

  const size_t XSZ = (size_t)kM * kD;     // 4096*512 elements
  const size_t WSZ = (size_t)kD * kHID;   // 512*512

  unsigned short* ws  = (unsigned short*)d_ws;
  unsigned short* Xq  = ws;
  unsigned short* Xk  = Xq  + XSZ;
  unsigned short* Xv  = Xk  + XSZ;
  unsigned short* WqT = Xv  + XSZ;
  unsigned short* WkT = WqT + WSZ;
  unsigned short* WvT = WkT + WSZ;
  unsigned short* WoT = WvT + WSZ;
  unsigned short* qp  = WoT + WSZ;
  unsigned short* kp  = qp  + XSZ;
  unsigned short* vpT = kp  + XSZ;
  unsigned short* cc  = vpT + XSZ;
  // total: 7*XSZ + 4*WSZ ~= 31.5 MB of ws

  const int cvtBlocks = (int)(XSZ / 256);
  cvt_bf16_kernel<<<cvtBlocks, 256, 0, stream>>>(q, Xq, (int)XSZ);
  cvt_bf16_kernel<<<cvtBlocks, 256, 0, stream>>>(k, Xk, (int)XSZ);
  cvt_bf16_kernel<<<cvtBlocks, 256, 0, stream>>>(v, Xv, (int)XSZ);

  const int trBlocks = (int)(WSZ / 256);
  transpose_cvt_kernel<<<trBlocks, 256, 0, stream>>>(Wq, WqT, kD, kHID);
  transpose_cvt_kernel<<<trBlocks, 256, 0, stream>>>(Wk, WkT, kD, kHID);
  transpose_cvt_kernel<<<trBlocks, 256, 0, stream>>>(Wv, WvT, kD, kHID);
  transpose_cvt_kernel<<<trBlocks, 256, 0, stream>>>(Wo, WoT, kHID, kD);

  dim3 ggrid(kM / 256, kHID / 64);        // (16, 8)
  gemm_bf16_wmma<<<ggrid, 256, 0, stream>>>(Xq, WqT, bq, qp,  kM, kHID, kD, 0);
  gemm_bf16_wmma<<<ggrid, 256, 0, stream>>>(Xk, WkT, bk, kp,  kM, kHID, kD, 0);
  gemm_bf16_wmma<<<ggrid, 256, 0, stream>>>(Xv, WvT, bv, vpT, kM, kHID, kD, 1);

  dim3 agrid(kS / 128, kB * kH);          // (16, 16)
  attn_wmma_kernel<<<agrid, 256, 0, stream>>>(qp, kp, vpT, cc);

  gemm_bf16_wmma<<<ggrid, 256, 0, stream>>>(cc, WoT, bo, d_out, kM, kD, kHID, 2);
}
